// InvariantPointAttentionV44_29197187678426
// MI455X (gfx1250) — matrix operand order repaired and checked by reading, not compiled, not run on hardware
//
#include <hip/hip_runtime.h>

// Problem constants
#define NN_TOK 768
#define C_DIM  768
#define H_HEAD 12
#define CH_DIM 64
#define P_PTS  4
#define DP_DIM 128
#define HP3    144
#define LDQ    96    // logits K: 64 scalar + 12 point, padded 76 -> 96 (mult of 32)
#define LDW    160   // point-out K: 144 padded -> 160 (mult of 32)
#define SCALE_ 0.125f
#define EPS_   1e-5f

typedef __attribute__((ext_vector_type(16))) _Float16 v16h;
typedef __attribute__((ext_vector_type(8)))  _Float16 v8h;
typedef __attribute__((ext_vector_type(8)))  float    v8f;
typedef __attribute__((ext_vector_type(2)))  float    v2f;

// ---------------------------------------------------------------------------
// f32 -> f16 plain convert (for the activation A matrix)
__global__ void convert_f16_kernel(const float* __restrict__ in,
                                   _Float16* __restrict__ out, int n) {
  int i = blockIdx.x * 256 + threadIdx.x;
  if (i < n) out[i] = (_Float16)in[i];
}

// f32 (K x N) -> f16 transposed (N x Kpad), zero padded in K
__global__ void convert_bt_kernel(const float* __restrict__ in,
                                  _Float16* __restrict__ out,
                                  int K, int Ncols, int Kpad) {
  int j = blockIdx.x;  // output row (original column)
  for (int kk = threadIdx.x; kk < Kpad; kk += 256)
    out[(size_t)j * Kpad + kk] =
        (kk < K) ? (_Float16)in[(size_t)kk * Ncols + j] : (_Float16)0.0f;
}

// Wpb (128 x 12) -> f32 transposed padded (16 x 128)
__global__ void convert_wpbT_kernel(const float* __restrict__ Wpb,
                                    float* __restrict__ WpbT) {
  for (int i = threadIdx.x; i < 16 * DP_DIM; i += 256) {
    int c = i / DP_DIM, k = i % DP_DIM;
    WpbT[c * DP_DIM + k] = (c < H_HEAD) ? Wpb[k * H_HEAD + c] : 0.0f;
  }
}

// ---------------------------------------------------------------------------
// fragment loaders (f16): A row-major MxK, Bt row-major NxK
__device__ __forceinline__ v16h load_frag(const _Float16* p0,
                                          const _Float16* p1) {
  v8h x0 = *(const v8h*)p0;
  v8h x1 = *(const v8h*)p1;
  v16h r;
#pragma unroll
  for (int i = 0; i < 8; ++i) { r[i] = x0[i]; r[i + 8] = x1[i]; }
  return r;
}

// ---------------------------------------------------------------------------
// Generic WMMA GEMM (16x16 tile / wave): D = A * Bt^T + bias + Cadd
__global__ void gemm_wmma_kernel(const _Float16* __restrict__ A, long long sA, int lda,
                                 const _Float16* __restrict__ Bt, long long sB, int ldb,
                                 const float* __restrict__ bias,
                                 const float* __restrict__ Cadd, int ldc,
                                 float* __restrict__ Dout, long long sD, int ldd,
                                 _Float16* __restrict__ DoutH, long long sDh, int lddh,
                                 int K) {
  const int lane = threadIdx.x;
  const int hi = lane >> 4, lm = lane & 15;
  const int bm = blockIdx.x * 16, bn = blockIdx.y * 16;
  const long long bz = blockIdx.z;
  const _Float16* arow = A + bz * sA + (size_t)(bm + lm) * lda;
  const _Float16* brow = Bt + bz * sB + (size_t)(bn + lm) * ldb;

  v8f acc;
#pragma unroll
  for (int i = 0; i < 8; ++i) acc[i] = 0.0f;

  for (int k = 0; k < K; k += 32) {
    v16h a = load_frag(arow + k + hi * 8, arow + k + hi * 8 + 16);
    v16h b = load_frag(brow + k + hi * 8, brow + k + hi * 8 + 16);
    acc = __builtin_amdgcn_wmma_f32_16x16x32_f16(false, a, false, b,
                                                 (short)0, acc, false, false);
  }

  const int col = bn + lm;
#pragma unroll
  for (int r = 0; r < 8; ++r) {
    const int row = bm + r + hi * 8;
    float v = acc[r];
    if (bias) v += bias[col];
    if (Cadd) v += Cadd[(size_t)row * ldc + col];
    if (Dout)  Dout[bz * sD + (size_t)row * ldd + col] = v;
    if (DoutH) DoutH[bz * sDh + (size_t)row * lddh + col] = (_Float16)v;
  }
}

// ---------------------------------------------------------------------------
// Register-blocked WMMA GEMM: 32x64 macro-tile per wave (2x4 tiles, 8 acc).
// Requires M%32==0, N%64==0, K%32==0.
__global__ void gemm_wmma_b24_kernel(const _Float16* __restrict__ A, int lda,
                                     const _Float16* __restrict__ Bt, int ldb,
                                     const float* __restrict__ bias,
                                     const float* __restrict__ Cadd, int ldc,
                                     float* __restrict__ Dout, int ldd,
                                     _Float16* __restrict__ DoutH, int lddh,
                                     int K) {
  const int lane = threadIdx.x;
  const int hi = lane >> 4, lm = lane & 15;
  const int bm = blockIdx.x * 32, bn = blockIdx.y * 64;

  const _Float16* arow[2];
  const _Float16* brow[4];
#pragma unroll
  for (int mt = 0; mt < 2; ++mt)
    arow[mt] = A + (size_t)(bm + mt * 16 + lm) * lda;
#pragma unroll
  for (int nt = 0; nt < 4; ++nt)
    brow[nt] = Bt + (size_t)(bn + nt * 16 + lm) * ldb;

  v8f acc[8];
#pragma unroll
  for (int t = 0; t < 8; ++t)
#pragma unroll
    for (int i = 0; i < 8; ++i) acc[t][i] = 0.0f;

  for (int k = 0; k < K; k += 32) {
    v16h af[2], bf[4];
#pragma unroll
    for (int mt = 0; mt < 2; ++mt)
      af[mt] = load_frag(arow[mt] + k + hi * 8, arow[mt] + k + hi * 8 + 16);
#pragma unroll
    for (int nt = 0; nt < 4; ++nt)
      bf[nt] = load_frag(brow[nt] + k + hi * 8, brow[nt] + k + hi * 8 + 16);
#pragma unroll
    for (int mt = 0; mt < 2; ++mt)
#pragma unroll
      for (int nt = 0; nt < 4; ++nt)
        acc[mt * 4 + nt] = __builtin_amdgcn_wmma_f32_16x16x32_f16(
            false, af[mt], false, bf[nt], (short)0, acc[mt * 4 + nt],
            false, false);
  }

#pragma unroll
  for (int mt = 0; mt < 2; ++mt)
#pragma unroll
    for (int nt = 0; nt < 4; ++nt) {
      const int col = bn + nt * 16 + lm;
#pragma unroll
      for (int r = 0; r < 8; ++r) {
        const int row = bm + mt * 16 + r + 8 * hi;
        float v = acc[mt * 4 + nt][r];
        if (bias) v += bias[col];
        if (Cadd) v += Cadd[(size_t)row * ldc + col];
        if (Dout)  Dout[(size_t)row * ldd + col] = v;
        if (DoutH) DoutH[(size_t)row * lddh + col] = (_Float16)v;
      }
    }
}

// ---------------------------------------------------------------------------
// Logits GEMM, blocked 32x64, batched over heads:
//   pb += Qc*Kc^T - 0.5*SCALE*(q2[row]+k2[col])     (in place)
__global__ void logits_wmma_b24_kernel(const _Float16* __restrict__ Qc,
                                       const _Float16* __restrict__ Kc,
                                       const float* __restrict__ q2,
                                       const float* __restrict__ k2,
                                       float* __restrict__ pb) {
  const int lane = threadIdx.x;
  const int hi = lane >> 4, lm = lane & 15;
  const int bm = blockIdx.x * 32, bn = blockIdx.y * 64;
  const long long h = blockIdx.z;

  const _Float16* Ah = Qc + h * (long long)NN_TOK * LDQ;
  const _Float16* Bh = Kc + h * (long long)NN_TOK * LDQ;
  const _Float16* arow[2];
  const _Float16* brow[4];
#pragma unroll
  for (int mt = 0; mt < 2; ++mt)
    arow[mt] = Ah + (size_t)(bm + mt * 16 + lm) * LDQ;
#pragma unroll
  for (int nt = 0; nt < 4; ++nt)
    brow[nt] = Bh + (size_t)(bn + nt * 16 + lm) * LDQ;

  v8f acc[8];
#pragma unroll
  for (int t = 0; t < 8; ++t)
#pragma unroll
    for (int i = 0; i < 8; ++i) acc[t][i] = 0.0f;

#pragma unroll
  for (int k = 0; k < LDQ; k += 32) {
    v16h af[2], bf[4];
#pragma unroll
    for (int mt = 0; mt < 2; ++mt)
      af[mt] = load_frag(arow[mt] + k + hi * 8, arow[mt] + k + hi * 8 + 16);
#pragma unroll
    for (int nt = 0; nt < 4; ++nt)
      bf[nt] = load_frag(brow[nt] + k + hi * 8, brow[nt] + k + hi * 8 + 16);
#pragma unroll
    for (int mt = 0; mt < 2; ++mt)
#pragma unroll
      for (int nt = 0; nt < 4; ++nt)
        acc[mt * 4 + nt] = __builtin_amdgcn_wmma_f32_16x16x32_f16(
            false, af[mt], false, bf[nt], (short)0, acc[mt * 4 + nt],
            false, false);
  }

  const size_t pbh = (size_t)h * NN_TOK * NN_TOK;
#pragma unroll
  for (int mt = 0; mt < 2; ++mt)
#pragma unroll
    for (int nt = 0; nt < 4; ++nt) {
      const int col = bn + nt * 16 + lm;
      const float k2c = k2[h * NN_TOK + col];
#pragma unroll
      for (int r = 0; r < 8; ++r) {
        const int row = bm + mt * 16 + r + 8 * hi;
        const size_t idx = pbh + (size_t)row * NN_TOK + col;
        pb[idx] = pb[idx] + acc[mt * 4 + nt][r]
                - 0.5f * SCALE_ * (q2[h * NN_TOK + row] + k2c);
      }
    }
}

// ---------------------------------------------------------------------------
// pair_bias on the matrix pipe (exact fp32, V_WMMA_F32_16X16X4_F32):
//   pb[h][nm] = sum_d pair[nm][d]*Wpb[d][h] + bpb[h]
// A tile = 16 consecutive nm rows (contiguous 8KB stream), B = WpbT (16x128).
__global__ void pair_bias_wmma_kernel(const float* __restrict__ pair,
                                      const float* __restrict__ WpbT,
                                      const float* __restrict__ bpb,
                                      float* __restrict__ pb) {
  const int lane = threadIdx.x & 31;
  const int hi = lane >> 4, lm = lane & 15;
  const long long wid = (long long)blockIdx.x * 8 + (threadIdx.x >> 5);
  const long long bm = wid * 16;  // 36864 tiles cover 768*768 rows
  const long long NNpix = (long long)NN_TOK * NN_TOK;

  const float* arow = pair + (size_t)(bm + lm) * DP_DIM;
  const float* brow = WpbT + (size_t)lm * DP_DIM;

  v8f acc;
#pragma unroll
  for (int i = 0; i < 8; ++i) acc[i] = 0.0f;

#pragma unroll
  for (int k = 0; k < DP_DIM; k += 4) {
    v2f a = *(const v2f*)(arow + k + hi * 2);
    v2f b = *(const v2f*)(brow + k + hi * 2);
    acc = __builtin_amdgcn_wmma_f32_16x16x4_f32(false, a, false, b,
                                                (short)0, acc, false, false);
  }

  const int col = lm;  // head index (cols 12..15 are padding)
  if (col < H_HEAD) {
    const float bias = bpb[col];
#pragma unroll
    for (int r = 0; r < 8; ++r) {
      const long long row = bm + r + 8 * hi;
      pb[(size_t)col * NNpix + row] = acc[r] + bias;
    }
  }
}

// ---------------------------------------------------------------------------
// Apply rotations/translations, build packed f16 operands:
//   Qc[h][n][0:64]=SCALE*q, [64:76]=SCALE*qg, [76:96]=0
//   Kc[h][n][0:64]=k,       [64:76]=kg,       [76:96]=0
//   VcT[h][cc][m]: cc 0:64=v, 64:76=vg, 76:80=0   (pre-transposed N x K)
//   q2[h][n], k2[h][n] = sum of squares of qg/kg
__global__ void rot_pack_kernel(const float* __restrict__ qbuf,
                                const float* __restrict__ kbuf,
                                const float* __restrict__ vbuf,
                                const float* __restrict__ qpb,
                                const float* __restrict__ kpb,
                                const float* __restrict__ vpb,
                                const float* __restrict__ rot,
                                const float* __restrict__ trans,
                                _Float16* __restrict__ Qc,
                                _Float16* __restrict__ Kc,
                                _Float16* __restrict__ VcT,
                                float* __restrict__ q2,
                                float* __restrict__ k2) {
  const int n = blockIdx.x, tid = threadIdx.x;
  __shared__ float sqg[HP3], skg[HP3];

  if (tid < HP3) {
    const int h = tid / 12, r12 = tid % 12, p = r12 / 3, e = r12 % 3;
    const float R0 = rot[n * 9 + 0 * 3 + e];
    const float R1 = rot[n * 9 + 1 * 3 + e];
    const float R2 = rot[n * 9 + 2 * 3 + e];
    const float T  = trans[n * 3 + e];
    const size_t base = (size_t)n * HP3 + h * 12 + p * 3;
    float qv = qpb[base + 0] * R0 + qpb[base + 1] * R1 + qpb[base + 2] * R2 + T;
    float kv = kpb[base + 0] * R0 + kpb[base + 1] * R1 + kpb[base + 2] * R2 + T;
    float vv = vpb[base + 0] * R0 + vpb[base + 1] * R1 + vpb[base + 2] * R2 + T;
    sqg[tid] = qv;
    skg[tid] = kv;
    const size_t ro = ((size_t)(h * NN_TOK + n)) * LDQ + 64 + p * 3 + e;
    Qc[ro] = (_Float16)(SCALE_ * qv);
    Kc[ro] = (_Float16)kv;
    VcT[((size_t)(h * 80 + 64 + p * 3 + e)) * NN_TOK + n] = (_Float16)vv;
  }
  __syncthreads();
  if (tid < 24) {
    const int h = tid % 12;
    const bool isQ = tid < 12;
    float s = 0.0f;
#pragma unroll
    for (int i = 0; i < 12; ++i) {
      float x = isQ ? sqg[h * 12 + i] : skg[h * 12 + i];
      s += x * x;
    }
    (isQ ? q2 : k2)[h * NN_TOK + n] = s;
  }
  for (int c = tid; c < C_DIM; c += 256) {
    const int h = c >> 6, j = c & 63;
    const size_t ro = ((size_t)(h * NN_TOK + n)) * LDQ + j;
    Qc[ro] = (_Float16)(SCALE_ * qbuf[(size_t)n * C_DIM + c]);
    Kc[ro] = (_Float16)kbuf[(size_t)n * C_DIM + c];
    VcT[((size_t)(h * 80 + j)) * NN_TOK + n] = (_Float16)vbuf[(size_t)n * C_DIM + c];
  }
  for (int i = tid; i < H_HEAD * 20; i += 256) {  // pad logits K 76..95
    const int h = i / 20, z = i % 20;
    const size_t ro = ((size_t)(h * NN_TOK + n)) * LDQ + 76 + z;
    Qc[ro] = (_Float16)0.0f;
    Kc[ro] = (_Float16)0.0f;
  }
  for (int i = tid; i < H_HEAD * 4; i += 256) {   // pad Vc cols 76..79
    const int h = i / 4, z = i % 4;
    VcT[((size_t)(h * 80 + 76 + z)) * NN_TOK + n] = (_Float16)0.0f;
  }
}

// ---------------------------------------------------------------------------
// Row softmax over logits (f32, in pb), write attention as f16.
__global__ void softmax_kernel(const float* __restrict__ logits,
                               _Float16* __restrict__ attnH) {
  __shared__ float red[256];
  const int tid = threadIdx.x;
  const float* L = logits + (size_t)blockIdx.x * NN_TOK;

  float m = -1e30f;
  for (int i = tid; i < NN_TOK; i += 256) m = fmaxf(m, L[i]);
  red[tid] = m;
  __syncthreads();
  for (int s = 128; s > 0; s >>= 1) {
    if (tid < s) red[tid] = fmaxf(red[tid], red[tid + s]);
    __syncthreads();
  }
  const float mx = red[0];
  __syncthreads();

  float s = 0.0f;
  for (int i = tid; i < NN_TOK; i += 256) s += __expf(L[i] - mx);
  red[tid] = s;
  __syncthreads();
  for (int st = 128; st > 0; st >>= 1) {
    if (tid < st) red[tid] += red[tid + st];
    __syncthreads();
  }
  const float inv = 1.0f / red[0];
  _Float16* O = attnH + (size_t)blockIdx.x * NN_TOK;
  for (int i = tid; i < NN_TOK; i += 256)
    O[i] = (_Float16)(__expf(L[i] - mx) * inv);
}

// ---------------------------------------------------------------------------
// Repack weighted-value output wv[n][h*80+cc] into
//   wsbuf[n][h*64+c] (f32, scalar part) and wpH[n][0..160) (f16, point part, padded)
__global__ void repack_wv_kernel(const float* __restrict__ wv,
                                 float* __restrict__ wsbuf,
                                 _Float16* __restrict__ wpH) {
  const int n = blockIdx.x, tid = threadIdx.x;
  for (int c = tid; c < C_DIM; c += 256)
    wsbuf[(size_t)n * C_DIM + c] = wv[(size_t)n * 960 + (c >> 6) * 80 + (c & 63)];
  for (int j = tid; j < LDW; j += 256)
    wpH[(size_t)n * LDW + j] =
        (j < HP3) ? (_Float16)wv[(size_t)n * 960 + (j / 12) * 80 + 64 + (j % 12)]
                  : (_Float16)0.0f;
}

// ---------------------------------------------------------------------------
// LayerNorm over last dim (768) of x, write final output.
__global__ void layernorm_kernel(const float* __restrict__ x,
                                 const float* __restrict__ gamma,
                                 const float* __restrict__ beta,
                                 float* __restrict__ out) {
  __shared__ float red[256];
  const int tid = threadIdx.x;
  const float* X = x + (size_t)blockIdx.x * C_DIM;

  float s = 0.0f;
  for (int i = tid; i < C_DIM; i += 256) s += X[i];
  red[tid] = s;
  __syncthreads();
  for (int st = 128; st > 0; st >>= 1) {
    if (tid < st) red[tid] += red[tid + st];
    __syncthreads();
  }
  const float mu = red[0] * (1.0f / C_DIM);
  __syncthreads();

  float v = 0.0f;
  for (int i = tid; i < C_DIM; i += 256) {
    float d = X[i] - mu;
    v += d * d;
  }
  red[tid] = v;
  __syncthreads();
  for (int st = 128; st > 0; st >>= 1) {
    if (tid < st) red[tid] += red[tid + st];
    __syncthreads();
  }
  const float inv = rsqrtf(red[0] * (1.0f / C_DIM) + EPS_);
  float* O = out + (size_t)blockIdx.x * C_DIM;
  for (int i = tid; i < C_DIM; i += 256)
    O[i] = gamma[i] * (X[i] - mu) * inv + beta[i];
}

// ---------------------------------------------------------------------------
extern "C" void kernel_launch(void* const* d_in, const int* in_sizes, int n_in,
                              void* d_out, int out_size, void* d_ws, size_t ws_size,
                              hipStream_t stream) {
  (void)in_sizes; (void)n_in; (void)out_size; (void)ws_size;
  const float* single = (const float*)d_in[0];
  const float* pair   = (const float*)d_in[1];
  const float* rot    = (const float*)d_in[2];
  const float* trans  = (const float*)d_in[3];
  const float* Wq  = (const float*)d_in[4];   const float* bq  = (const float*)d_in[5];
  const float* Wk  = (const float*)d_in[6];   const float* bk  = (const float*)d_in[7];
  const float* Wv  = (const float*)d_in[8];   const float* bv  = (const float*)d_in[9];
  const float* Wpb = (const float*)d_in[10];  const float* bpb = (const float*)d_in[11];
  const float* Wqp = (const float*)d_in[12];  const float* bqp = (const float*)d_in[13];
  const float* Wkp = (const float*)d_in[14];  const float* bkp = (const float*)d_in[15];
  const float* Wvp = (const float*)d_in[16];  const float* bvp = (const float*)d_in[17];
  const float* Wo  = (const float*)d_in[18];  const float* bo  = (const float*)d_in[19];
  const float* Wpo = (const float*)d_in[20];  const float* bpo = (const float*)d_in[21];
  const float* gamma = (const float*)d_in[22];
  const float* beta  = (const float*)d_in[23];
  float* out = (float*)d_out;

  const size_t NC = (size_t)NN_TOK * C_DIM;          // 589824
  const size_t NNpix = (size_t)NN_TOK * NN_TOK;      // 589824

  // Workspace arena (256B aligned slices)
  char* p = (char*)d_ws;
  auto alloc = [&](size_t bytes) -> void* {
    void* r = (void*)p;
    p += (bytes + 255) & ~(size_t)255;
    return r;
  };
  float* qbuf  = (float*)alloc(NC * 4);
  float* kbuf  = (float*)alloc(NC * 4);
  float* vbuf  = (float*)alloc(NC * 4);
  float* qpb   = (float*)alloc((size_t)NN_TOK * HP3 * 4);
  float* kpb   = (float*)alloc((size_t)NN_TOK * HP3 * 4);
  float* vpb   = (float*)alloc((size_t)NN_TOK * HP3 * 4);
  float* q2    = (float*)alloc((size_t)H_HEAD * NN_TOK * 4);
  float* k2    = (float*)alloc((size_t)H_HEAD * NN_TOK * 4);
  float* WpbT  = (float*)alloc((size_t)16 * DP_DIM * 4);
  float* pb    = (float*)alloc((size_t)H_HEAD * NNpix * 4);      // logits in-place
  float* wv    = (float*)alloc((size_t)NN_TOK * 960 * 4);
  float* wsbuf = (float*)alloc(NC * 4);
  float* xbuf  = (float*)alloc(NC * 4);
  _Float16* singleH = (_Float16*)alloc(NC * 2);
  _Float16* WqT  = (_Float16*)alloc(NC * 2);
  _Float16* WkT  = (_Float16*)alloc(NC * 2);
  _Float16* WvT  = (_Float16*)alloc(NC * 2);
  _Float16* WoT  = (_Float16*)alloc(NC * 2);
  _Float16* WqpT = (_Float16*)alloc((size_t)HP3 * C_DIM * 2);
  _Float16* WkpT = (_Float16*)alloc((size_t)HP3 * C_DIM * 2);
  _Float16* WvpT = (_Float16*)alloc((size_t)HP3 * C_DIM * 2);
  _Float16* WpoT = (_Float16*)alloc((size_t)C_DIM * LDW * 2);
  _Float16* Qc   = (_Float16*)alloc((size_t)H_HEAD * NN_TOK * LDQ * 2);
  _Float16* Kc   = (_Float16*)alloc((size_t)H_HEAD * NN_TOK * LDQ * 2);
  _Float16* VcT  = (_Float16*)alloc((size_t)H_HEAD * 80 * NN_TOK * 2);
  _Float16* attnH = (_Float16*)alloc((size_t)H_HEAD * NNpix * 2);
  _Float16* wpH  = (_Float16*)alloc((size_t)NN_TOK * LDW * 2);
  _Float16* tH   = (_Float16*)alloc(NC * 2);

  // 1) convert activations + weights (weights pre-transposed N x Kpad)
  convert_f16_kernel<<<(NC + 255) / 256, 256, 0, stream>>>(single, singleH, (int)NC);
  convert_bt_kernel<<<C_DIM, 256, 0, stream>>>(Wq,  WqT,  C_DIM, C_DIM, C_DIM);
  convert_bt_kernel<<<C_DIM, 256, 0, stream>>>(Wk,  WkT,  C_DIM, C_DIM, C_DIM);
  convert_bt_kernel<<<C_DIM, 256, 0, stream>>>(Wv,  WvT,  C_DIM, C_DIM, C_DIM);
  convert_bt_kernel<<<C_DIM, 256, 0, stream>>>(Wo,  WoT,  C_DIM, C_DIM, C_DIM);
  convert_bt_kernel<<<HP3,   256, 0, stream>>>(Wqp, WqpT, C_DIM, HP3,   C_DIM);
  convert_bt_kernel<<<HP3,   256, 0, stream>>>(Wkp, WkpT, C_DIM, HP3,   C_DIM);
  convert_bt_kernel<<<HP3,   256, 0, stream>>>(Wvp, WvpT, C_DIM, HP3,   C_DIM);
  convert_bt_kernel<<<C_DIM, 256, 0, stream>>>(Wpo, WpoT, HP3,   C_DIM, LDW);
  convert_wpbT_kernel<<<1, 256, 0, stream>>>(Wpb, WpbT);

  // 2) q/k/v (blocked) and point projections (generic, N=144)
  dim3 gB(24, 12, 1), g69(48, 9, 1), blk(32);
  gemm_wmma_b24_kernel<<<gB, blk, 0, stream>>>(singleH, C_DIM, WqT, C_DIM, bq,
      nullptr, 0, qbuf, C_DIM, nullptr, 0, C_DIM);
  gemm_wmma_b24_kernel<<<gB, blk, 0, stream>>>(singleH, C_DIM, WkT, C_DIM, bk,
      nullptr, 0, kbuf, C_DIM, nullptr, 0, C_DIM);
  gemm_wmma_b24_kernel<<<gB, blk, 0, stream>>>(singleH, C_DIM, WvT, C_DIM, bv,
      nullptr, 0, vbuf, C_DIM, nullptr, 0, C_DIM);
  gemm_wmma_kernel<<<g69, blk, 0, stream>>>(singleH, 0, C_DIM, WqpT, 0, C_DIM, bqp,
      nullptr, 0, qpb, 0, HP3, nullptr, 0, 0, C_DIM);
  gemm_wmma_kernel<<<g69, blk, 0, stream>>>(singleH, 0, C_DIM, WkpT, 0, C_DIM, bkp,
      nullptr, 0, kpb, 0, HP3, nullptr, 0, 0, C_DIM);
  gemm_wmma_kernel<<<g69, blk, 0, stream>>>(singleH, 0, C_DIM, WvpT, 0, C_DIM, bvp,
      nullptr, 0, vpb, 0, HP3, nullptr, 0, 0, C_DIM);

  // 3) rotate points, build packed Qc/Kc/VcT + q2/k2
  rot_pack_kernel<<<NN_TOK, 256, 0, stream>>>(qbuf, kbuf, vbuf, qpb, kpb, vpb,
                                              rot, trans, Qc, Kc, VcT, q2, k2);

  // 4) pair bias on the matrix pipe (302 MB stream -> memory bound)
  pair_bias_wmma_kernel<<<4608, 256, 0, stream>>>(pair, WpbT, bpb, pb);

  // 5) fused logits (blocked): pb += [SCALE*q|SCALE*qg].[k|kg]^T - 0.5*SCALE*(q2+k2)
  dim3 gl(24, 12, H_HEAD);
  logits_wmma_b24_kernel<<<gl, blk, 0, stream>>>(Qc, Kc, q2, k2, pb);

  // 6) softmax rows -> f16 attention
  softmax_kernel<<<H_HEAD * NN_TOK, 256, 0, stream>>>(pb, attnH);

  // 7) weighted values: per head  attn (768x768) @ [v|vg] (768x80)
  dim3 gw(48, 5, H_HEAD);
  gemm_wmma_kernel<<<gw, blk, 0, stream>>>(attnH, (long long)NNpix, NN_TOK,
      VcT, (long long)80 * NN_TOK, NN_TOK, nullptr, nullptr, 0,
      wv, 80, 960, nullptr, 0, 0, NN_TOK);

  // 8) split scalar / point parts
  repack_wv_kernel<<<NN_TOK, 256, 0, stream>>>(wv, wsbuf, wpH);

  // 9) t = ws + wp @ Wpo + bpo  (f16 out for next GEMM, blocked)
  gemm_wmma_b24_kernel<<<gB, blk, 0, stream>>>(wpH, LDW, WpoT, LDW, bpo,
      wsbuf, C_DIM, nullptr, 0, tH, C_DIM, LDW);

  // 10) x = single + t @ Wo + bo  (blocked)
  gemm_wmma_b24_kernel<<<gB, blk, 0, stream>>>(tH, C_DIM, WoT, C_DIM, bo,
      single, C_DIM, xbuf, C_DIM, nullptr, 0, C_DIM);

  // 11) LayerNorm -> output
  layernorm_kernel<<<NN_TOK, 256, 0, stream>>>(xbuf, gamma, beta, out);
}
